// SymbolicOperator_77464030150842
// MI455X (gfx1250) — matrix operand order repaired and checked by reading, not compiled, not run on hardware
//
#include <hip/hip_runtime.h>
#include <hip/hip_bf16.h>
#include <math.h>

// ---------------- CDNA5 vector types for WMMA ----------------
typedef __attribute__((ext_vector_type(16))) _Float16 v16h;
typedef __attribute__((ext_vector_type(8)))  _Float16 v8h;
typedef __attribute__((ext_vector_type(8)))  float    v8f;
typedef __attribute__((ext_vector_type(4)))  int      v4i;

#define B_   256
#define LW   10     // words actually processed: instructions[1:-1]
#define LL   50     // L = true_actions[1:] length
#define V_   1000
#define P_   200
#define PK   224    // P padded to multiple of 32 for WMMA K
#define KD_  128
#define H_   384    // 3*KD
#define H3   1152   // 3*H
#define STEPS_ 3

// =====================================================================
// WMMA GEMM, one wave computes a 16x64 strip (4 N-tiles), A fragment
// reused across the 4 WMMAs:  C[M x ldc](f32) = A[M x lda](f16) * W[N x ldw](f16)^T
// Grid supplies exactly (M/16)*(N/64) waves (blockDim=256 => 8 waves/block).
// A-fragment (16x32 f16): lanes 0-15 -> rows M=0..15, halves {K0..K7, K16..K23};
//                         lanes 16-31 -> same rows,    halves {K8..K15,K24..K31}.
// B-fragment (32x16 f16): lane&15 = column N; lanes<16 hold K=0..7,16..23, lanes>=16 K=8..15,24..31.
// D (16x16 f32, 8 VGPRs): n = lane&15, m = (lane>=16?8:0)+i.
// =====================================================================
__global__ void wmma_gemm_nt4(const _Float16* __restrict__ A, int lda,
                              const _Float16* __restrict__ W, int ldw,
                              float* __restrict__ C, int ldc,
                              int ntilesN4, int K) {
  int gw   = (blockIdx.x * blockDim.x + threadIdx.x) >> 5;
  int lane = threadIdx.x & 31;
  int tm = gw / ntilesN4, tn4 = gw % ntilesN4;
  int hsel = lane >> 4;        // 0 for lanes 0-15, 1 for lanes 16-31
  int nr   = lane & 15;
  const _Float16* arow  = A + (size_t)(tm * 16 + nr) * lda;
  const _Float16* wrow0 = W + (size_t)(tn4 * 64 + nr) * ldw;
  v8f acc[4] = {{}, {}, {}, {}};
  for (int k0 = 0; k0 < K; k0 += 32) {
    v8h a0 = *(const v8h*)(arow + k0 + hsel * 8);
    v8h a1 = *(const v8h*)(arow + k0 + 16 + hsel * 8);
    if (k0 + 32 < K) __builtin_prefetch(arow + k0 + 32, 0, 3);  // near-scope prefetch
    v16h a;
#pragma unroll
    for (int i = 0; i < 8; ++i) { a[i] = a0[i]; a[8 + i] = a1[i]; }
#pragma unroll
    for (int j = 0; j < 4; ++j) {
      const _Float16* wr = wrow0 + (size_t)(j * 16) * ldw + k0 + hsel * 8;
      v8h b0 = *(const v8h*)(wr);
      v8h b1 = *(const v8h*)(wr + 16);
      v16h b;
#pragma unroll
      for (int i = 0; i < 8; ++i) { b[i] = b0[i]; b[8 + i] = b1[i]; }
      acc[j] = __builtin_amdgcn_wmma_f32_16x16x32_f16(false, a, false, b,
                                                      (short)0, acc[j], false, false);
    }
  }
  int mbase = tm * 16 + hsel * 8;
#pragma unroll
  for (int j = 0; j < 4; ++j)
#pragma unroll
    for (int i = 0; i < 8; ++i)
      C[(size_t)(mbase + i) * ldc + tn4 * 64 + j * 16 + nr] = acc[j][i];
}

// =====================================================================
// Fused attention-logits kernel: read_p and write_p projections vs keys.
// keys (64x128 f16 = 16 KB) staged into LDS once per block via CDNA5
// async global->LDS copy (fallback: plain copy), then WMMA B-fragments
// come from LDS (ds_load_b128). 32 waves total: sel = read/write, tm = M tile.
// Each wave computes a full 16x64 logits strip.
// =====================================================================
__global__ void attn_logits_k(const _Float16* __restrict__ h_h,     // B x H (f16)
                              const _Float16* __restrict__ keys_h,  // 64 x 128 (f16)
                              float* __restrict__ logits_r,         // B x 64
                              float* __restrict__ logits_w) {       // B x 64
  __shared__ __align__(16) _Float16 keys_lds[64 * KD_];   // 16 KB
  int tid = threadIdx.x;
  // ---- stage keys into LDS: 16384 B = 4 passes of 256 threads x 16 B ----
#if __has_builtin(__builtin_amdgcn_global_load_async_to_lds_b128)
#pragma unroll
  for (int i = 0; i < 4; ++i) {
    int off = i * 4096 + tid * 16;
    __builtin_amdgcn_global_load_async_to_lds_b128(
        (__attribute__((address_space(1))) v4i*)((const char*)keys_h + off),
        (__attribute__((address_space(3))) v4i*)((char*)keys_lds + off),
        0, 0);
  }
#if __has_builtin(__builtin_amdgcn_s_wait_asynccnt)
  __builtin_amdgcn_s_wait_asynccnt(0);
#else
  asm volatile("s_wait_asynccnt 0x0" ::: "memory");
#endif
#else
#pragma unroll
  for (int i = 0; i < 4; ++i) {
    int off = i * 4096 + tid * 16;
    *(v8h*)((char*)keys_lds + off) = *(const v8h*)((const char*)keys_h + off);
  }
#endif
  __syncthreads();

  int gw   = blockIdx.x * 8 + (tid >> 5);   // 0..31
  int lane = tid & 31;
  int tm  = gw & 15;                         // M tile (256 rows -> 16 tiles)
  int sel = gw >> 4;                         // 0 = read_p, 1 = write_p
  int hsel = lane >> 4;
  int nr   = lane & 15;
  const _Float16* arow = h_h + (size_t)(tm * 16 + nr) * H_ + (1 + sel) * KD_;
  v8f acc[4] = {{}, {}, {}, {}};
  for (int k0 = 0; k0 < KD_; k0 += 32) {
    v8h a0 = *(const v8h*)(arow + k0 + hsel * 8);
    v8h a1 = *(const v8h*)(arow + k0 + 16 + hsel * 8);
    v16h a;
#pragma unroll
    for (int i = 0; i < 8; ++i) { a[i] = a0[i]; a[8 + i] = a1[i]; }
#pragma unroll
    for (int j = 0; j < 4; ++j) {
      const _Float16* kr = keys_lds + (size_t)(j * 16 + nr) * KD_ + k0 + hsel * 8;
      v8h b0 = *(const v8h*)(kr);          // ds_load_b128
      v8h b1 = *(const v8h*)(kr + 16);     // ds_load_b128
      v16h b;
#pragma unroll
      for (int i = 0; i < 8; ++i) { b[i] = b0[i]; b[8 + i] = b1[i]; }
      acc[j] = __builtin_amdgcn_wmma_f32_16x16x32_f16(false, a, false, b,
                                                      (short)0, acc[j], false, false);
    }
  }
  float* C = sel ? logits_w : logits_r;
  int mbase = tm * 16 + hsel * 8;
#pragma unroll
  for (int j = 0; j < 4; ++j)
#pragma unroll
    for (int i = 0; i < 8; ++i)
      C[(size_t)(mbase + i) * 64 + j * 16 + nr] = acc[j][i];
}

// -------- f32 -> f16 convert with zero padding (dst is Np x Kp) --------
__global__ void pad_f16_k(const float* __restrict__ src, _Float16* __restrict__ dst,
                          int Np, int Kp, int N, int K) {
  int idx = blockIdx.x * blockDim.x + threadIdx.x;
  if (idx >= Np * Kp) return;
  int n = idx / Kp, k = idx % Kp;
  dst[idx] = (n < N && k < K) ? (_Float16)src[(size_t)n * K + k] : (_Float16)0.f;
}

// -------- h0 = tile(scratch_keys[0], (B, 3)), plus f16 shadow --------
__global__ void init_h_k(const float* __restrict__ scratch_keys,
                         float* __restrict__ h, _Float16* __restrict__ h_h) {
  int idx = blockIdx.x * blockDim.x + threadIdx.x;   // B*H
  float v = scratch_keys[idx % KD_];                 // row 0 of pe
  h[idx] = v;
  h_h[idx] = (_Float16)v;
}

// -------- sv0 = tile(init_value, (B, L, 1)) --------
__global__ void init_sv_k(const float* __restrict__ init_value, float* __restrict__ sv) {
  size_t idx = (size_t)blockIdx.x * 256 + threadIdx.x;
  sv[idx] = init_value[idx % V_];
}

// -------- per-word: gate softmax, word ids, padded f16 program gather --------
__global__ void word_prep_k(const int* __restrict__ instructions, int w,
                            const float* __restrict__ gate_emb,
                            const float* __restrict__ program_emb,
                            int* __restrict__ words, float* __restrict__ g0,
                            float* __restrict__ g1, _Float16* __restrict__ prog_h) {
  int b = threadIdx.x;
  int word = instructions[(w + 1) * B_ + b];
  words[b] = word;
  float a0 = gate_emb[word * 2 + 0], a1 = gate_emb[word * 2 + 1];
  float m = fmaxf(a0, a1);
  float e0 = __expf(a0 - m), e1 = __expf(a1 - m);
  float inv = 1.f / (e0 + e1);
  g0[b] = e0 * inv; g1[b] = e1 * inv;
  for (int k = 0; k < PK; ++k)
    prog_h[b * PK + k] = (k < P_) ? (_Float16)program_emb[(size_t)word * P_ + k]
                                  : (_Float16)0.f;
}

// -------- softmax over 50 valid columns of a 256x64 logits buffer --------
__global__ void softmax50_k(const float* __restrict__ logits, float* __restrict__ attn) {
  int b = blockIdx.x * blockDim.x + threadIdx.x;
  if (b >= B_) return;
  const float* row = logits + b * 64;
  float m = -1e30f;
  for (int l = 0; l < LL; ++l) m = fmaxf(m, row[l]);
  float s = 0.f;
  for (int l = 0; l < LL; ++l) s += __expf(row[l] - m);
  float inv = 1.f / s;
  for (int l = 0; l < LL; ++l) attn[b * LL + l] = __expf(row[l] - m) * inv;
}

// -------- rv[b,v] = sum_l attn_r[b,l] * sv[b,l,v] --------
__global__ void read_value_k(const float* __restrict__ sv,
                             const float* __restrict__ attn_r, float* __restrict__ rv) {
  int v = blockIdx.x * 256 + threadIdx.x;
  int b = blockIdx.y;
  if (v >= V_) return;
  const float* svb = sv + (size_t)b * LL * V_;
  const float* ar  = attn_r + b * LL;
  float acc = 0.f;
#pragma unroll 5
  for (int l = 0; l < LL; ++l) acc = fmaf(ar[l], svb[l * V_ + v], acc);
  rv[b * V_ + v] = acc;
}

// -------- sv = wm*new_value + (1-wm)*sv, new_value fused (primitive gather + rv) -----
__global__ void sv_update_k(float* __restrict__ sv, const float* __restrict__ attn_w,
                            const float* __restrict__ rv,
                            const float* __restrict__ primitive_emb,
                            const int* __restrict__ words,
                            const float* __restrict__ g0, const float* __restrict__ g1) {
  int v = blockIdx.x * 256 + threadIdx.x;
  int l = blockIdx.y, b = blockIdx.z;
  if (v >= V_) return;
  int word = words[b];
  float nv  = g0[b] * primitive_emb[(size_t)word * V_ + v] + g1[b] * rv[b * V_ + v];
  float wmv = attn_w[b * LL + l];
  size_t idx = (size_t)b * LL * V_ + (size_t)l * V_ + v;
  sv[idx] = wmv * nv + (1.f - wmv) * sv[idx];
}

// -------- GRU pointwise + gated hidden blend; refresh f16 shadow of h --------
__global__ void gru_pointwise_k(const float* __restrict__ gi, const float* __restrict__ gh,
                                const float* __restrict__ b_ih, const float* __restrict__ b_hh,
                                const float* __restrict__ g0, const float* __restrict__ g1,
                                float* __restrict__ h, _Float16* __restrict__ h_h) {
  int idx = blockIdx.x * blockDim.x + threadIdx.x;   // B*H
  int b = idx / H_, j = idx % H_;
  size_t base = (size_t)b * H3;
  float ir = gi[base + j]          + b_ih[j];
  float hr = gh[base + j]          + b_hh[j];
  float iz = gi[base + H_ + j]     + b_ih[H_ + j];
  float hz = gh[base + H_ + j]     + b_hh[H_ + j];
  float in_ = gi[base + 2 * H_ + j] + b_ih[2 * H_ + j];
  float hn  = gh[base + 2 * H_ + j] + b_hh[2 * H_ + j];
  float r = 1.f / (1.f + __expf(-(ir + hr)));
  float z = 1.f / (1.f + __expf(-(iz + hz)));
  float n = tanhf(in_ + r * hn);
  float hv = h[idx];
  float hnew = (1.f - z) * n + z * hv;
  float out = g0[b] * hv + g1[b] * hnew;
  h[idx] = out;
  h_h[idx] = (_Float16)out;
}

// -------- pass 1 of log_softmax: row max + log-sum-exp per (b,l) --------
__global__ void lse_k(const float* __restrict__ sv, float* __restrict__ rowm,
                      float* __restrict__ rowlse) {
  __shared__ float red[128];
  int bl = blockIdx.x;                      // b*LL + l
  const float* row = sv + (size_t)bl * V_;
  int tid = threadIdx.x;
  float m = -1e30f;
  for (int v = tid; v < V_; v += 128) m = fmaxf(m, row[v]);
  red[tid] = m; __syncthreads();
  for (int s = 64; s > 0; s >>= 1) { if (tid < s) red[tid] = fmaxf(red[tid], red[tid + s]); __syncthreads(); }
  m = red[0]; __syncthreads();
  float s = 0.f;
  for (int v = tid; v < V_; v += 128) s += __expf(row[v] - m);
  red[tid] = s; __syncthreads();
  for (int st = 64; st > 0; st >>= 1) { if (tid < st) red[tid] += red[tid + st]; __syncthreads(); }
  if (tid == 0) { rowm[bl] = m; rowlse[bl] = logf(red[0]); }
}

// -------- pass 2: out[l,v,b] = sv[b,l,v] - max - lse (coalesced writes) --------
__global__ void write_actions_k(const float* __restrict__ sv, const float* __restrict__ rowm,
                                const float* __restrict__ rowlse, float* __restrict__ out) {
  int b = threadIdx.x;     // 256
  int v = blockIdx.x;      // 1000
  int l = blockIdx.y;      // 50
  int bl = b * LL + l;
  float val = sv[(size_t)b * LL * V_ + (size_t)l * V_ + v] - rowm[bl] - rowlse[bl];
  out[((size_t)l * V_ + v) * B_ + b] = val;
}

// -------- acts.T : out2[b,l] = true_actions[l+1, b] --------
__global__ void write_acts_k(const int* __restrict__ true_actions, float* __restrict__ out2) {
  int b = threadIdx.x;     // 256
  int l = blockIdx.x;      // 50
  out2[b * LL + l] = (float)true_actions[(l + 1) * B_ + b];
}

// =====================================================================
extern "C" void kernel_launch(void* const* d_in, const int* in_sizes, int n_in,
                              void* d_out, int out_size, void* d_ws, size_t ws_size,
                              hipStream_t stream) {
  const float* gate_emb      = (const float*)d_in[0];   // (1000, 2)
  const float* program_emb   = (const float*)d_in[1];   // (1000, 200)
  const float* primitive_emb = (const float*)d_in[2];   // (1000, 1000)
  const float* scratch_keys  = (const float*)d_in[3];   // (50, 128)
  const float* init_value    = (const float*)d_in[4];   // (1000,)
  const float* w_ih          = (const float*)d_in[5];   // (1152, 200)
  const float* w_hh          = (const float*)d_in[6];   // (1152, 384)
  const float* b_ih          = (const float*)d_in[7];   // (1152,)
  const float* b_hh          = (const float*)d_in[8];   // (1152,)
  const int*   instructions  = (const int*)d_in[9];     // (12, 256)
  const int*   true_actions  = (const int*)d_in[10];    // (51, 256)
  float* out = (float*)d_out;

  // ---- workspace carve (aligned 256B) ----
  char* p = (char*)d_ws;
  auto alloc = [&](size_t bytes) -> void* {
    void* r = (void*)p; p += (bytes + 255) & ~(size_t)255; return r;
  };
  float*    sv      = (float*)   alloc((size_t)B_ * LL * V_ * 4);  // 51.2 MB
  float*    h       = (float*)   alloc((size_t)B_ * H_ * 4);
  _Float16* h_h     = (_Float16*)alloc((size_t)B_ * H_ * 2);
  _Float16* w_ih_h  = (_Float16*)alloc((size_t)H3 * PK * 2);
  _Float16* w_hh_h  = (_Float16*)alloc((size_t)H3 * H_ * 2);
  _Float16* keys_h  = (_Float16*)alloc((size_t)64 * KD_ * 2);
  _Float16* prog_h  = (_Float16*)alloc((size_t)B_ * PK * 2);
  float*    gi      = (float*)   alloc((size_t)B_ * H3 * 4);
  float*    gh      = (float*)   alloc((size_t)B_ * H3 * 4);
  float*    logits_r= (float*)   alloc((size_t)B_ * 64 * 4);
  float*    logits_w= (float*)   alloc((size_t)B_ * 64 * 4);
  float*    attn_r  = (float*)   alloc((size_t)B_ * LL * 4);
  float*    attn_w  = (float*)   alloc((size_t)B_ * LL * 4);
  float*    rv      = (float*)   alloc((size_t)B_ * V_ * 4);
  float*    g0      = (float*)   alloc(B_ * 4);
  float*    g1      = (float*)   alloc(B_ * 4);
  int*      words   = (int*)     alloc(B_ * 4);
  float*    rowm    = (float*)   alloc((size_t)B_ * LL * 4);
  float*    rowlse  = (float*)   alloc((size_t)B_ * LL * 4);
  (void)ws_size; (void)in_sizes; (void)n_in; (void)out_size;

  // ---- one-time (per call) init: f16 weight shadows + state ----
  pad_f16_k<<<(H3 * PK + 255) / 256, 256, 0, stream>>>(w_ih, w_ih_h, H3, PK, H3, P_);
  pad_f16_k<<<(H3 * H_ + 255) / 256, 256, 0, stream>>>(w_hh, w_hh_h, H3, H_, H3, H_);
  pad_f16_k<<<(64 * KD_ + 255) / 256, 256, 0, stream>>>(scratch_keys, keys_h, 64, KD_, LL, KD_);
  init_h_k<<<(B_ * H_) / 256, 256, 0, stream>>>(scratch_keys, h, h_h);
  init_sv_k<<<(B_ * LL * V_) / 256, 256, 0, stream>>>(init_value, sv);

  for (int w = 0; w < LW; ++w) {
    word_prep_k<<<1, 256, 0, stream>>>(instructions, w, gate_emb, program_emb,
                                       words, g0, g1, prog_h);
    // gi = program @ w_ih^T : constant across the 3 inner steps -> hoisted.
    // M=256 (16 tiles) x N=1152 (18 strip-tiles of 64) -> 288 waves = 36 blocks.
    wmma_gemm_nt4<<<36, 256, 0, stream>>>(prog_h, PK, w_ih_h, PK, gi, H3, 18, PK);

    for (int s = 0; s < STEPS_; ++s) {
      // fused read/write attention logits vs LDS-staged keys (async copy-in).
      attn_logits_k<<<4, 256, 0, stream>>>(h_h, keys_h, logits_r, logits_w);
      softmax50_k<<<1, 256, 0, stream>>>(logits_r, attn_r);
      softmax50_k<<<1, 256, 0, stream>>>(logits_w, attn_w);

      read_value_k<<<dim3(4, B_), 256, 0, stream>>>(sv, attn_r, rv);
      sv_update_k<<<dim3(4, LL, B_), 256, 0, stream>>>(sv, attn_w, rv, primitive_emb,
                                                       words, g0, g1);
      // gh = h @ w_hh^T : M=256, N=1152 -> 288 waves = 36 blocks, K=384.
      wmma_gemm_nt4<<<36, 256, 0, stream>>>(h_h, H_, w_hh_h, H_, gh, H3, 18, H_);
      gru_pointwise_k<<<(B_ * H_) / 256, 256, 0, stream>>>(gi, gh, b_ih, b_hh,
                                                           g0, g1, h, h_h);
    }
  }

  // ---- final log_softmax over V, transposed to (L, V, B); then acts.T ----
  lse_k<<<B_ * LL, 128, 0, stream>>>(sv, rowm, rowlse);
  write_actions_k<<<dim3(V_, LL), B_, 0, stream>>>(sv, rowm, rowlse, out);
  write_acts_k<<<LL, B_, 0, stream>>>(true_actions, out + (size_t)LL * V_ * B_);
}
